// NodeAttention_SP_35055523070518
// MI455X (gfx1250) — matrix operand order repaired and checked by reading, not compiled, not run on hardware
//
#include <hip/hip_runtime.h>
#include <hip/hip_bf16.h>

typedef __attribute__((ext_vector_type(16))) __bf16 v16bf;
typedef __attribute__((ext_vector_type(8)))  __bf16 v8bf;
typedef __attribute__((ext_vector_type(8)))  float  v8f;
typedef __attribute__((ext_vector_type(4)))  float  v4f;

#define N_NODES 50000
#define DEG     32
#define F_IN    512
#define F_OUT   256
#define MTILE   32      // rows per workgroup in the GEMM

// ---------------------------------------------------------------------------
// Kernel 0: convert W [512,256] f32 -> bf16, swizzled into the WMMA B-fragment
// layout: Wsw[((kt*256 + col)*2 + half)*16 + j] = W[kt*32 + (j>>3)*16 + half*8 + (j&7), col]
// so each lane's 16 K-values are contiguous (one 32B load, wave-coalesced 1KB).
// ---------------------------------------------------------------------------
__global__ __launch_bounds__(256)
void wconv_kernel(const float* __restrict__ W, __bf16* __restrict__ Wsw) {
    int o = blockIdx.x * blockDim.x + threadIdx.x;
    if (o >= F_IN * F_OUT) return;
    int j  = o & 15;
    int h  = (o >> 4) & 1;
    int c  = (o >> 5) & (F_OUT - 1);
    int kt = o >> 13;
    int k  = kt * 32 + ((j & 8) << 1) + h * 8 + (j & 7);
    Wsw[o] = (__bf16)W[k * F_OUT + c];
}

// ---------------------------------------------------------------------------
// Kernel 1: seq = x @ W via V_WMMA_F32_16X16X32_BF16.
// Workgroup = 256 threads (8 waves), computes a 32-row x 256-col strip.
// Each wave: 2 row-blocks x 2 col-tiles = 4 accumulators; B fragments are
// reused across the two row-blocks (halves L2 traffic for W vs 16-row tiles).
// A strip (32 x 512) staged in LDS as bf16 (32 KB), converted from f32.
// ---------------------------------------------------------------------------
__global__ __launch_bounds__(256)
void gat_gemm_kernel(const float* __restrict__ x, const __bf16* __restrict__ Wsw,
                     float* __restrict__ seq) {
    __shared__ __align__(16) __bf16 Ash[MTILE * F_IN];   // 32 KB
    const int tid  = threadIdx.x;
    const int row0 = blockIdx.x * MTILE;

    // stage A: MTILE rows x 512 K, coalesced f32 loads, convert to bf16.
    // Clamp source row for the final partial block (no OOB reads).
    for (int i = tid; i < MTILE * F_IN; i += 256) {
        int r = i >> 9;                       // i / F_IN
        int k = i & (F_IN - 1);
        int grow = row0 + r;
        if (grow > N_NODES - 1) grow = N_NODES - 1;
        Ash[i] = (__bf16)x[(size_t)grow * F_IN + k];
    }
    __syncthreads();

    const int lane = tid & 31;
    const int wave = tid >> 5;
    const int l16  = (lane >= 16) ? 1 : 0;
    const int arow = lane & 15;
    const int c0   = wave * 32;
    const int colA = c0 + (lane & 15);

    v8f acc00 = {}, acc01 = {};   // row-block 0, col tiles 0/1
    v8f acc10 = {}, acc11 = {};   // row-block 1, col tiles 0/1

    for (int kt = 0; kt < F_IN / 32; ++kt) {
        const int kk = kt * 32;
        // A fragment per ISA layout: lane<16 -> K {kk..+7, kk+16..+23},
        //                            lane>=16 -> K {kk+8..+15, kk+24..+31}
        const int kb = kk + l16 * 8;
        v16bf a0, a1;
        {
            const v8bf* ap = (const v8bf*)&Ash[arow * F_IN + kb];
            v8bf lo = ap[0], hi = ap[2];
            #pragma unroll
            for (int i = 0; i < 8; ++i) { a0[i] = lo[i]; a0[8 + i] = hi[i]; }
        }
        {
            const v8bf* ap = (const v8bf*)&Ash[(16 + arow) * F_IN + kb];
            v8bf lo = ap[0], hi = ap[2];
            #pragma unroll
            for (int i = 0; i < 8; ++i) { a1[i] = lo[i]; a1[8 + i] = hi[i]; }
        }

        // B fragments: contiguous 32B per lane from swizzled W (1KB/wave)
        v16bf b0 = *(const v16bf*)&Wsw[(((size_t)kt * F_OUT + colA)      * 2 + l16) * 16];
        v16bf b1 = *(const v16bf*)&Wsw[(((size_t)kt * F_OUT + colA + 16) * 2 + l16) * 16];

        acc00 = __builtin_amdgcn_wmma_f32_16x16x32_bf16(false, a0, false, b0,
                                                        (short)0, acc00, false, false);
        acc01 = __builtin_amdgcn_wmma_f32_16x16x32_bf16(false, a0, false, b1,
                                                        (short)0, acc01, false, false);
        acc10 = __builtin_amdgcn_wmma_f32_16x16x32_bf16(false, a1, false, b0,
                                                        (short)0, acc10, false, false);
        acc11 = __builtin_amdgcn_wmma_f32_16x16x32_bf16(false, a1, false, b1,
                                                        (short)0, acc11, false, false);
    }

    // D layout: VGPR r, lane l -> row = rb*16 + r + 8*(l>=16), col = lane&15
    #pragma unroll
    for (int r = 0; r < 8; ++r) {
        int row = row0 + l16 * 8 + r;
        if (row < N_NODES) {
            seq[(size_t)row * F_OUT + colA]      = acc00[r];
            seq[(size_t)row * F_OUT + colA + 16] = acc01[r];
        }
        int row2 = row + 16;
        if (row2 < N_NODES) {
            seq[(size_t)row2 * F_OUT + colA]      = acc10[r];
            seq[(size_t)row2 * F_OUT + colA + 16] = acc11[r];
        }
    }
}

// ---------------------------------------------------------------------------
// Kernel 2: f1/f2 = seq @ a1_w / a2_w + biases. One wave (32 lanes) per node,
// 8 f32 per lane, wave32 shuffle reduction.
// ---------------------------------------------------------------------------
__global__ __launch_bounds__(256)
void attn_scores_kernel(const float* __restrict__ seq,
                        const float* __restrict__ a1w, const float* __restrict__ a1b,
                        const float* __restrict__ a2w, const float* __restrict__ a2b,
                        float* __restrict__ f1, float* __restrict__ f2) {
    const int lane = threadIdx.x & 31;
    const int wave = threadIdx.x >> 5;
    const int node = blockIdx.x * 8 + wave;
    if (node >= N_NODES) return;
    const float* s = seq + (size_t)node * F_OUT + lane * 8;
    float d1 = 0.f, d2 = 0.f;
    #pragma unroll
    for (int i = 0; i < 8; ++i) {
        float v = s[i];
        d1 += v * a1w[lane * 8 + i];
        d2 += v * a2w[lane * 8 + i];
    }
    #pragma unroll
    for (int off = 16; off >= 1; off >>= 1) {
        d1 += __shfl_xor(d1, off, 32);
        d2 += __shfl_xor(d2, off, 32);
    }
    if (lane == 0) {
        f1[node] = d1 + a1b[0];
        f2[node] = d2 + a2b[0];
    }
}

// ---------------------------------------------------------------------------
// Kernel 3: per-node edge logits -> leaky_relu -> softmax over the 32
// contiguous edges -> weighted gather of seq rows (L2 resident) -> +bias ->
// ELU. 4 nodes per 256-thread block; even waves compute the 32-wide softmax
// for their node; 64 threads/node aggregate with float4 (b128 gathers).
// ---------------------------------------------------------------------------
__global__ __launch_bounds__(256)
void attn_aggregate_kernel(const float* __restrict__ seq,
                           const float* __restrict__ f1, const float* __restrict__ f2,
                           const float* __restrict__ bias_vals,
                           const int* __restrict__ edge_col,
                           const float* __restrict__ bias_out,
                           float* __restrict__ out) {
    __shared__ float coef[4][DEG];
    __shared__ int   ncol[4][DEG];
    const int tid = threadIdx.x;
    const int g   = tid >> 6;            // node group 0..3
    const int t   = tid & 63;            // thread within group
    const int node = blockIdx.x * 4 + g;

    if (t < DEG) {                       // whole even waves: full 32-lane EXEC
        const int   eidx = node * DEG + t;
        const int   col  = edge_col[eidx];
        const float bv   = bias_vals[eidx];
        float e = bv * f1[node] + bv * f2[col];
        e = (e >= 0.f) ? e : 0.2f * e;                   // leaky_relu(0.2)
        float m = e;
        #pragma unroll
        for (int off = 16; off >= 1; off >>= 1)
            m = fmaxf(m, __shfl_xor(m, off, 32));
        float ee = __expf(e - m);
        float s = ee;
        #pragma unroll
        for (int off = 16; off >= 1; off >>= 1)
            s += __shfl_xor(s, off, 32);
        coef[g][t] = ee / fmaxf(s, 1e-30f);
        ncol[g][t] = col;
    }
    __syncthreads();

    v4f acc = {};
    #pragma unroll 4
    for (int k = 0; k < DEG; ++k) {
        const v4f sv = *(const v4f*)&seq[(size_t)ncol[g][k] * F_OUT + 4 * t];
        acc += coef[g][k] * sv;
    }
    v4f b = *(const v4f*)&bias_out[4 * t];
    v4f v = acc + b;
    v4f r;
    #pragma unroll
    for (int i = 0; i < 4; ++i)
        r[i] = (v[i] > 0.f) ? v[i] : (__expf(v[i]) - 1.f);
    *(v4f*)&out[(size_t)node * F_OUT + 4 * t] = r;
}

// ---------------------------------------------------------------------------
extern "C" void kernel_launch(void* const* d_in, const int* in_sizes, int n_in,
                              void* d_out, int out_size, void* d_ws, size_t ws_size,
                              hipStream_t stream) {
    const float* x         = (const float*)d_in[0];
    const float* W         = (const float*)d_in[1];
    const float* a1w       = (const float*)d_in[2];
    const float* a1b       = (const float*)d_in[3];
    const float* a2w       = (const float*)d_in[4];
    const float* a2b       = (const float*)d_in[5];
    const float* bias_out  = (const float*)d_in[6];
    const float* bias_vals = (const float*)d_in[7];
    const int*   edge_col  = (const int*)d_in[9];
    float* out = (float*)d_out;

    char* ws = (char*)d_ws;
    __bf16* Wsw = (__bf16*)ws;                                        // 256 KB
    float*  seq = (float*)(ws + (256 << 10));                         // 51.2 MB
    float*  f1  = (float*)(ws + (256 << 10) + (size_t)N_NODES * F_OUT * 4);
    float*  f2  = f1 + N_NODES;

    wconv_kernel<<<(F_IN * F_OUT + 255) / 256, 256, 0, stream>>>(W, Wsw);
    gat_gemm_kernel<<<(N_NODES + MTILE - 1) / MTILE, 256, 0, stream>>>(x, Wsw, seq);
    attn_scores_kernel<<<N_NODES / 8, 256, 0, stream>>>(seq, a1w, a1b, a2w, a2b, f1, f2);
    attn_aggregate_kernel<<<N_NODES / 4, 256, 0, stream>>>(seq, f1, f2, bias_vals,
                                                           edge_col, bias_out, out);
}
